// AJRNN_66297115181588
// MI455X (gfx1250) — matrix operand order repaired and checked by reading, not compiled, not run on hardware
//
#include <hip/hip_runtime.h>
#include <hip/hip_bf16.h>

// ---------------------------------------------------------------------------
// AJ-RNN (2-layer LSTM w/ missing-value imputation) for gfx1250 / MI455X.
//
// v5: v4 + explicit 2-stage software pipelining of fragment loads.
// The recurrence runs at ~1 wave/SIMD (latency-bound, 32 WGs), so loads for
// chunk kc+1 are issued into a second register buffer before the WMMAs of
// chunk kc, turning full s_wait_loadcnt 0x0 stalls into partial waits.
//
// - Weights fp32->bf16 + swizzled once into per-lane B-fragment layout;
//   L2-resident (6.3 MB << 192 MB). All offsets are immediates (24-bit IOFFSET).
// - h1/h2 kept in bf16 A-fragment layout, written directly by the pointwise
//   stage (inverse fragment map), double-buffered across timesteps.
// - Layer 2 / pred: zero LDS, zero barriers. Layer 1 stages only the 16x64
//   imputed input tile once per step (branchless select).
// - 3 kernels per timestep, 768-kernel chain graph-captured by the harness.
// ---------------------------------------------------------------------------

typedef __attribute__((ext_vector_type(16))) __bf16 v16bf;
typedef __attribute__((ext_vector_type(8)))  float  v8f;

#define B_ 128
#define T_ 256
#define D_ 64
#define H_ 512
#define MISSING_ 128.0f

// byte strides in the swizzled layouts (one v16bf fragment slot = 32 B)
#define A_KC_STRIDE   8192      // 8 mblks * 32 lanes * 32 B
#define B_KC_STRIDE   131072    // 128 nblocks * 32 lanes * 32 B (N = 2048)
#define B_GATE_STRIDE 32768     // 32 nblocks  * 32 lanes * 32 B
#define P_KC_STRIDE   4096      // 4 nblocks   * 32 lanes * 32 B (N = 64)

// --- fragment K-index maps (CDNA5 ISA 7.12.2, 16-bit A 16x32 & B 32x16) ----
__device__ __forceinline__ int kbaseA(int half, int v) {
  // A (16x32): half0 covers K 0-7 & 16-23, half1 covers K 8-15 & 24-31
  return (v < 4) ? (half * 8 + v * 2) : (16 + half * 8 + (v - 4) * 2);
}
__device__ __forceinline__ int kbaseB(int half, int v) {
  // B (32x16): half0 covers K 0-7 & 16-23, half1 covers K 8-15 & 24-31
  return (v < 4) ? (half * 8 + v * 2) : (16 + half * 8 + (v - 4) * 2);
}

// fast gate nonlinearities: v_exp_f32 + v_rcp_f32 (ample accuracy vs bf16 data)
__device__ __forceinline__ float fast_sigmoid(float x) {
  return __builtin_amdgcn_rcpf(1.0f + __expf(-x));
}
__device__ __forceinline__ float fast_tanh(float x) {
  return 1.0f - 2.0f * __builtin_amdgcn_rcpf(__expf(2.0f * x) + 1.0f);
}

// ---------------------------------------------------------------------------
// One-time weight conversion: fp32 [Ktot x N] (split across two sources at
// row ksplit) -> bf16 swizzled B-fragment layout:
//   dst[ ((kc*nBlocks + nb)*32 + lane)*16 + i ]
// ---------------------------------------------------------------------------
__global__ void __launch_bounds__(256)
swizzle_w(const float* __restrict__ W0, const float* __restrict__ W1,
          int ksplit, int Ktot, int N, __bf16* __restrict__ dst) {
  int tid = blockIdx.x * blockDim.x + threadIdx.x;
  if (tid >= Ktot * N) return;
  int i       = tid & 15;
  int lane    = (tid >> 4) & 31;
  int nBlocks = N >> 4;
  int nb      = (tid >> 9) % nBlocks;
  int kc      = tid / (nBlocks << 9);
  int n       = nb * 16 + (lane & 15);
  int half    = lane >> 4;
  int k       = kc * 32 + kbaseB(half, i >> 1) + (i & 1);
  float val = (k < ksplit) ? W0[k * N + n] : W1[(k - ksplit) * N + n];
  dst[tid] = (__bf16)val;
}

// load the 4 gate B fragments for chunk KC into pipeline slot S
#define LOAD_B(S, KC)                                                          \
  {                                                                            \
    bq[S][0] = *(const v16bf*)(wp + (KC) * B_KC_STRIDE + 0 * B_GATE_STRIDE);   \
    bq[S][1] = *(const v16bf*)(wp + (KC) * B_KC_STRIDE + 1 * B_GATE_STRIDE);   \
    bq[S][2] = *(const v16bf*)(wp + (KC) * B_KC_STRIDE + 2 * B_GATE_STRIDE);   \
    bq[S][3] = *(const v16bf*)(wp + (KC) * B_KC_STRIDE + 3 * B_GATE_STRIDE);   \
  }

// 4-gate WMMA step consuming pipeline slot S with A fragment AF
#define GATE_WMMA(AF, S)                                                       \
  {                                                                            \
    acc0 = __builtin_amdgcn_wmma_f32_16x16x32_bf16(false, AF, false, bq[S][0], \
                                                   (short)0, acc0, false, false);\
    acc1 = __builtin_amdgcn_wmma_f32_16x16x32_bf16(false, AF, false, bq[S][1], \
                                                   (short)0, acc1, false, false);\
    acc2 = __builtin_amdgcn_wmma_f32_16x16x32_bf16(false, AF, false, bq[S][2], \
                                                   (short)0, acc2, false, false);\
    acc3 = __builtin_amdgcn_wmma_f32_16x16x32_bf16(false, AF, false, bq[S][3], \
                                                   (short)0, acc3, false, false);\
  }

// Shared LSTM pointwise: bias + gates, update c (fp32), write h in swizzled
// bf16 (and optionally plain fp32 for the last_cell output).
__device__ __forceinline__ void
lstm_pointwise(v8f acc0, v8f acc1, v8f acc2, v8f acc3,
               const float* __restrict__ bias, float* __restrict__ c_state,
               __bf16* __restrict__ hswz, float* __restrict__ hplain,
               int mblk, int nb, int m16, int half) {
  const int hunit = nb * 16 + m16;                 // C/D column = lane%16
  const float bi = bias[0 * H_ + hunit];
  const float bf = bias[1 * H_ + hunit];
  const float bg = bias[2 * H_ + hunit];
  const float bo = bias[3 * H_ + hunit];
  // inverse A-fragment map for column k = hunit (constant per lane)
  const int kcH = hunit >> 5, kkH = hunit & 31;
  const int p   = kkH & 1,    q   = kkH >> 1;
  const int halfA = (q >> 2) & 1;
  const int vv    = (q & 3) + ((q >> 3) << 2);
  const int sbase = (((kcH * 8 + mblk) * 32 + halfA * 16) << 4) + 2 * vv + p;
#pragma unroll
  for (int r = 0; r < 8; ++r) {
    int m    = r + 8 * half;                       // C/D row within tile
    int brow = mblk * 16 + m;
    float ig = fast_sigmoid(acc0[r] + bi);
    float fg = fast_sigmoid(acc1[r] + bf);
    float gg = fast_tanh(acc2[r] + bg);
    float og = fast_sigmoid(acc3[r] + bo);
    float cold = c_state[brow * H_ + hunit];
    float cnew = fg * cold + ig * gg;
    c_state[brow * H_ + hunit] = cnew;
    float hnew = og * fast_tanh(cnew);
    hswz[sbase + (m << 4)] = (__bf16)hnew;
    if (hplain) hplain[brow * H_ + hunit] = hnew;
  }
}

// ---------------------------------------------------------------------------
// Layer 1: z = [cur(x,pred) | h1_in] @ Wcat + b0 ; K = 64 + 512 = 18 chunks.
// cur tile (16x64) staged once in LDS with branchless imputation.
// ---------------------------------------------------------------------------
__global__ void __launch_bounds__(256)
lstm_layer1(const float* __restrict__ x, const float* __restrict__ pred,
            const __bf16* __restrict__ h1s_in, const __bf16* __restrict__ wB,
            const float* __restrict__ bias, float* __restrict__ c_state,
            __bf16* __restrict__ h1s_out, int t) {
  __shared__ __bf16 curTile[16 * 64];
  const int mblk = blockIdx.x & 7;
  const int ngrp = blockIdx.x >> 3;
  const int wave = threadIdx.x >> 5;
  const int lane = threadIdx.x & 31;
  const int nb   = ngrp * 8 + wave;               // h-unit block 0..31
  const int m16  = lane & 15;
  const int half = lane >> 4;

  // ---- stage cur = where(t>0 & x==MISSING, pred, x) once (4 elem/thread) --
  {
    int idx  = threadIdx.x * 4;                   // contiguous in d
    int mm   = idx >> 6;
    int d0   = idx & 63;
    int brow = mblk * 16 + mm;
#pragma unroll
    for (int e = 0; e < 4; ++e) {
      int d = d0 + e;
      float xv = x[(brow * T_ + t) * D_ + d];     // coalesced b128
      float pv = pred[brow * D_ + d];             // unconditional (buffer valid)
      bool imp = (t > 0) && (xv == MISSING_);     // -> v_cndmask, no branches
      curTile[mm * 64 + d] = (__bf16)(imp ? pv : xv);
    }
  }
  __syncthreads();

  v8f acc0 = {}, acc1 = {}, acc2 = {}, acc3 = {};
  const char* wp = (const char*)wB + (size_t)(nb * 32 + lane) * 32;
  const char* ap = (const char*)h1s_in + (size_t)(mblk * 32 + lane) * 32;

  // LDS-sourced A fragments for the input chunks 0..1
  v16bf afX[2];
#pragma unroll
  for (int kc = 0; kc < 2; ++kc)
#pragma unroll
    for (int v = 0; v < 8; ++v) {
      int kb = kc * 32 + kbaseA(half, v);
      afX[kc][2 * v]     = curTile[m16 * 64 + kb];
      afX[kc][2 * v + 1] = curTile[m16 * 64 + kb + 1];
    }

  // ---- 2-stage software pipeline over 18 chunks ----
  v16bf bq[2][4];
  v16bf aq[2];
  LOAD_B(0, 0);
#pragma unroll
  for (int kc = 0; kc < 18; ++kc) {
    const int cur = kc & 1, nxt = cur ^ 1;
    if (kc + 1 < 18) {                            // prefetch next chunk
      LOAD_B(nxt, kc + 1);
      if (kc + 1 >= 2)
        aq[nxt] = *(const v16bf*)(ap + (kc + 1 - 2) * A_KC_STRIDE);
    }
    v16bf af = (kc < 2) ? afX[kc] : aq[cur];
    GATE_WMMA(af, cur);
  }

  lstm_pointwise(acc0, acc1, acc2, acc3, bias, c_state, h1s_out, nullptr,
                 mblk, nb, m16, half);
}

// ---------------------------------------------------------------------------
// Layer 2: z = [h1_new | h2_in] @ Wcat + b1 ; K = 1024 = 32 chunks. No LDS.
// Also writes plain fp32 h2 for the last_cell output.
// ---------------------------------------------------------------------------
__global__ void __launch_bounds__(256)
lstm_layer2(const __bf16* __restrict__ h1s, const __bf16* __restrict__ h2s_in,
            const __bf16* __restrict__ wB, const float* __restrict__ bias,
            float* __restrict__ c_state, __bf16* __restrict__ h2s_out,
            float* __restrict__ h2_plain) {
  const int mblk = blockIdx.x & 7;
  const int ngrp = blockIdx.x >> 3;
  const int wave = threadIdx.x >> 5;
  const int lane = threadIdx.x & 31;
  const int nb   = ngrp * 8 + wave;
  const int m16  = lane & 15;
  const int half = lane >> 4;

  v8f acc0 = {}, acc1 = {}, acc2 = {}, acc3 = {};
  const char* wp  = (const char*)wB + (size_t)(nb * 32 + lane) * 32;
  const char* ap1 = (const char*)h1s    + (size_t)(mblk * 32 + lane) * 32;
  const char* ap2 = (const char*)h2s_in + (size_t)(mblk * 32 + lane) * 32;

  // ---- 2-stage software pipeline over 32 chunks (16 from h1, 16 from h2) --
  v16bf bq[2][4];
  v16bf aq[2];
  LOAD_B(0, 0);
  aq[0] = *(const v16bf*)(ap1 + 0 * A_KC_STRIDE);
#pragma unroll
  for (int kc = 0; kc < 32; ++kc) {
    const int cur = kc & 1, nxt = cur ^ 1;
    if (kc + 1 < 32) {                            // prefetch next chunk
      LOAD_B(nxt, kc + 1);
      aq[nxt] = (kc + 1 < 16)
                    ? *(const v16bf*)(ap1 + (kc + 1) * A_KC_STRIDE)
                    : *(const v16bf*)(ap2 + (kc + 1 - 16) * A_KC_STRIDE);
    }
    GATE_WMMA(aq[cur], cur);
  }

  lstm_pointwise(acc0, acc1, acc2, acc3, bias, c_state, h2s_out, h2_plain,
                 mblk, nb, m16, half);
}

// ---------------------------------------------------------------------------
// pred = h2 @ W + bias [128 x 64]: output row t (t < T-1) and imputation
// source for step t+1. A side reads the swizzled h2 directly.
// ---------------------------------------------------------------------------
__global__ void __launch_bounds__(256)
pred_kernel(const __bf16* __restrict__ h2s, const __bf16* __restrict__ wP,
            const float* __restrict__ bias, float* __restrict__ pred,
            float* __restrict__ out, int t) {
  const int wave = threadIdx.x >> 5;
  const int lane = threadIdx.x & 31;
  const int w    = blockIdx.x * 8 + wave;         // 0..31
  const int mblk = w >> 2;                        // 0..7
  const int nb   = w & 3;                         // 0..3 (D=64 -> 4 blocks)
  const int m16  = lane & 15;
  const int half = lane >> 4;

  v8f acc = {};
  const char* ap = (const char*)h2s + (size_t)(mblk * 32 + lane) * 32;
  const char* wp = (const char*)wP  + (size_t)(nb * 32 + lane) * 32;

  // 2-stage pipeline
  v16bf aq[2], bp[2];
  aq[0] = *(const v16bf*)(ap);
  bp[0] = *(const v16bf*)(wp);
#pragma unroll
  for (int kc = 0; kc < 16; ++kc) {
    const int cur = kc & 1, nxt = cur ^ 1;
    if (kc + 1 < 16) {
      aq[nxt] = *(const v16bf*)(ap + (kc + 1) * A_KC_STRIDE);
      bp[nxt] = *(const v16bf*)(wp + (kc + 1) * P_KC_STRIDE);
    }
    acc = __builtin_amdgcn_wmma_f32_16x16x32_bf16(false, aq[cur], false, bp[cur],
                                                  (short)0, acc, false, false);
  }

  const int d  = nb * 16 + m16;
  const float bd = bias[d];
#pragma unroll
  for (int r = 0; r < 8; ++r) {
    int brow  = mblk * 16 + r + 8 * half;
    float val = acc[r] + bd;
    pred[brow * D_ + d] = val;                               // imputation src
    if (t < T_ - 1)
      out[((long)brow * (T_ - 1) + t) * D_ + d] = val;       // prediction row
  }
}

__global__ void __launch_bounds__(256)
copy_last(const float* __restrict__ h2, float* __restrict__ dst) {
  int i = blockIdx.x * blockDim.x + threadIdx.x;
  dst[i] = h2[i];
}

// ---------------------------------------------------------------------------
extern "C" void kernel_launch(void* const* d_in, const int* in_sizes, int n_in,
                              void* d_out, int out_size, void* d_ws, size_t ws_size,
                              hipStream_t stream) {
  const float* x    = (const float*)d_in[0];
  const float* k0   = (const float*)d_in[1];
  const float* r0   = (const float*)d_in[2];
  const float* b0   = (const float*)d_in[3];
  const float* k1   = (const float*)d_in[4];
  const float* r1   = (const float*)d_in[5];
  const float* b1   = (const float*)d_in[6];
  const float* W    = (const float*)d_in[7];
  const float* bias = (const float*)d_in[8];
  float* out = (float*)d_out;
  (void)in_sizes; (void)n_in; (void)out_size; (void)ws_size;

  // ---- workspace layout (zeroed region first) ----
  const size_t S = (size_t)B_ * H_;               // 65536 elems per state buf
  char* p = (char*)d_ws;
  float*  c1      = (float*)p;                    p += S * 4;       // fp32
  float*  c2      = (float*)p;                    p += S * 4;       // fp32
  float*  predbuf = (float*)p;                    p += (size_t)B_ * D_ * 4;
  __bf16* h1s[2]  = { (__bf16*)p, (__bf16*)(p + S * 2) };  p += 2 * S * 2;
  __bf16* h2s[2]  = { (__bf16*)p, (__bf16*)(p + S * 2) };  p += 2 * S * 2;
  size_t zero_bytes = (size_t)(p - (char*)d_ws);  // c1,c2,pred,h swz buffers
  float*  h2_plain = (float*)p;                   p += S * 4;
  __bf16* wB1 = (__bf16*)p;                       p += (size_t)576 * 2048 * 2;
  __bf16* wB2 = (__bf16*)p;                       p += (size_t)1024 * 2048 * 2;
  __bf16* wP  = (__bf16*)p;

  hipMemsetAsync(d_ws, 0, zero_bytes, stream);

  // ---- one-time weight convert + swizzle (L2-resident thereafter) ----
  {
    int n1 = 576 * 2048;                          // concat(K0;R0) along K
    swizzle_w<<<(n1 + 255) / 256, 256, 0, stream>>>(k0, r0, 64, 576, 2048, wB1);
    int n2 = 1024 * 2048;                         // concat(K1;R1) along K
    swizzle_w<<<(n2 + 255) / 256, 256, 0, stream>>>(k1, r1, 512, 1024, 2048, wB2);
    int n3 = 512 * 64;                            // prediction kernel W
    swizzle_w<<<(n3 + 255) / 256, 256, 0, stream>>>(W, W, 512, 512, 64, wP);
  }

  // ---- sequential recurrence: 3 kernels per timestep ----
  for (int t = 0; t < T_; ++t) {
    const __bf16* h1in = h1s[t & 1];  __bf16* h1out = h1s[(t + 1) & 1];
    const __bf16* h2in = h2s[t & 1];  __bf16* h2out = h2s[(t + 1) & 1];
    lstm_layer1<<<32, 256, 0, stream>>>(x, predbuf, h1in, wB1, b0, c1, h1out, t);
    lstm_layer2<<<32, 256, 0, stream>>>(h1out, h2in, wB2, b1, c2, h2out, h2_plain);
    pred_kernel<<<4, 256, 0, stream>>>(h2out, wP, bias, predbuf, out, t);
  }

  // last_cell = h2 after final step
  copy_last<<<(int)(S / 256), 256, 0, stream>>>(h2_plain,
                                                out + (size_t)B_ * (T_ - 1) * D_);
}
